// LookupConv2d_78048145703020
// MI455X (gfx1250) — matrix operand order repaired and checked by reading, not compiled, not run on hardware
//
#include <hip/hip_runtime.h>

// ---------------------------------------------------------------------------
// LookupConv2d for MI455X (gfx1250): bf16 WMMA implicit GEMM.
//   M = spatial pixels, N = Cout, K = (kh*3+kw)*128 + cin  (1152)
//   B-tile (weights) double-buffered in LDS via async global->LDS DMA.
// ---------------------------------------------------------------------------

#ifndef __has_builtin
#define __has_builtin(x) 0
#endif

#if __has_builtin(__builtin_amdgcn_global_load_async_to_lds_b128)
#define USE_ASYNC_LDS 1
#else
#define USE_ASYNC_LDS 0
#endif

typedef __bf16 bf16_t;
typedef __attribute__((ext_vector_type(16))) __bf16        v16bf;
typedef __attribute__((ext_vector_type(8)))  float         v8f;
typedef __attribute__((ext_vector_type(4)))  unsigned int  u32x4;
typedef int i32x4g __attribute__((vector_size(16)));   // matches builtin param

struct Q2 { u32x4 lo, hi; };   // 32 bytes -> bit_cast to v16bf

#define B_N    32
#define CIN    128
#define HW     56
#define COUT   256
#define KDIM   1152            // 9 * 128
#define NCHUNK 36              // KDIM / 32
#define BROW   40              // padded LDS row (elements): 32 + 8 (bank spread)

// ---------------------------------------------------------------------------
// Kernel 1: synthesize bf16 weight matrix  Wbf[Cout][1152], k = r*128 + c
// ---------------------------------------------------------------------------
__global__ __launch_bounds__(128) void build_weights_kernel(
    const float* __restrict__ dict, const float* __restrict__ coeff,
    const int* __restrict__ idx, bf16_t* __restrict__ wbf) {
  const int o = blockIdx.x;     // output channel 0..255
  const int c = threadIdx.x;    // input channel 0..127
  float cs[8]; int is[8];
#pragma unroll
  for (int s = 0; s < 8; ++s) { cs[s] = coeff[o * 8 + s]; is[s] = idx[o * 8 + s]; }
#pragma unroll
  for (int r = 0; r < 9; ++r) {
    float acc = 0.f;
#pragma unroll
    for (int s = 0; s < 8; ++s)
      acc += cs[s] * dict[(size_t)is[s] * KDIM + c * 9 + r];
    wbf[(size_t)o * KDIM + r * CIN + c] = (bf16_t)acc;
  }
}

// ---------------------------------------------------------------------------
// Kernel 2: x NCHW f32 -> xb NHWC bf16 (25.7 MB; L2-resident afterwards).
// ---------------------------------------------------------------------------
__global__ __launch_bounds__(128) void convert_x_kernel(
    const float* __restrict__ x, bf16_t* __restrict__ xb) {
  const int by = blockIdx.x;          // b*56 + y
  const int b  = by / HW, y = by % HW;
  const int c  = threadIdx.x;         // 0..127
  const float* src = x + (((size_t)(b * CIN + c) * HW) + y) * HW;
  bf16_t* dst = xb + (size_t)by * HW * CIN + c;
  for (int xp = 0; xp < HW; ++xp)
    dst[(size_t)xp * CIN] = (bf16_t)src[xp];
}

// ---------------------------------------------------------------------------
// Helpers
// ---------------------------------------------------------------------------
static __device__ __forceinline__ void wait_async0() {
#if USE_ASYNC_LDS
#if __has_builtin(__builtin_amdgcn_s_wait_asynccnt)
  __builtin_amdgcn_s_wait_asynccnt(0);
#else
  asm volatile("s_wait_asynccnt 0" ::: "memory");
#endif
#endif
}

// Stage one K-chunk of weights (256 Cout x 32 K bf16 = 16KB) into LDS.
// 1024 x 16B chunks; 256 threads x 4.  LDS rows padded to BROW elements.
static __device__ __forceinline__ void stage_b(const bf16_t* __restrict__ wbf,
                                               int kc, bf16_t* bs, int tid) {
#pragma unroll
  for (int i = 0; i < 4; ++i) {
    const int f   = tid + i * 256;      // 16B chunk id, 0..1023
    const int o   = f >> 2;             // Cout row
    const int sub = (f & 3) * 8;        // element offset within the 32-K row
    const bf16_t* gp = wbf + (size_t)o * KDIM + kc * 32 + sub;
    bf16_t*       lp = bs + o * BROW + sub;
#if USE_ASYNC_LDS
    __builtin_amdgcn_global_load_async_to_lds_b128((i32x4g*)gp, (i32x4g*)lp,
                                                   0, 0);
#else
    *reinterpret_cast<u32x4*>(lp) = *reinterpret_cast<const u32x4*>(gp);
#endif
  }
}

// A fragment: 16x32 bf16 A tile, documented gfx1250 layout.
//   lane<16 : v0..3 = K0..7   (bytes 0..15),  v4..7 = K16..23 (bytes 32..47)
//   lane>=16: v0..3 = K8..15  (bytes 16..31), v4..7 = K24..31 (bytes 48..63)
static __device__ __forceinline__ v16bf load_a_frag(const bf16_t* p, bool valid,
                                                    int half) {
  u32x4 lo = {0u, 0u, 0u, 0u};
  u32x4 hi = {0u, 0u, 0u, 0u};
  if (valid) {
    lo = *reinterpret_cast<const u32x4*>(p + half * 8);
    hi = *reinterpret_cast<const u32x4*>(p + 16 + half * 8);
  }
  Q2 q{lo, hi};
  return __builtin_bit_cast(v16bf, q);
}

// ---------------------------------------------------------------------------
// Kernel 3: conv as implicit GEMM with v_wmma_f32_16x16x32_bf16.
//   block = one (b,h) output row, w padded 56->64.  8 waves = 2(M) x 4(N).
//   wave tile = 32 px x 64 Cout = 8 accumulators; per chunk: 8 WMMA.
//   Weights double-buffered in LDS (async DMA overlaps WMMA).
// ---------------------------------------------------------------------------
__global__ __launch_bounds__(256) void lookup_conv_kernel(
    const bf16_t* __restrict__ xb, const bf16_t* __restrict__ wbf,
    float* __restrict__ out) {
  __shared__ alignas(128) bf16_t Bs[2][COUT * BROW];   // 2 x 20KB

  const int tid   = threadIdx.x;
  const int lane  = tid & 31;
  const int wave  = tid >> 5;
  const int waveM = wave & 1;                 // 2 M-rows of 32 px
  const int waveN = wave >> 1;                // 4 N-cols of 64 Cout
  const int bh = blockIdx.x;                  // b*56 + h
  const int b  = bh / HW, h = bh % HW;
  const int half = lane >> 4;                 // 0 / 1 (K-half select)
  const int lan  = lane & 15;

  // halo rows form a contiguous invalid prefix/suffix of the chunk range
  const int kcB = (h == 0)      ? 12 : 0;
  const int kcE = (h == HW - 1) ? 24 : NCHUNK;

  v8f acc[2][4] = {};

  stage_b(wbf, kcB, &Bs[0][0], tid);          // prologue DMA

  for (int kc = kcB; kc < kcE; ++kc) {
    const int buf = (kc - kcB) & 1;
    wait_async0();                            // my DMA for Bs[buf] landed
    __syncthreads();                          // everyone's DMA landed
    if (kc + 1 < kcE) stage_b(wbf, kc + 1, &Bs[buf ^ 1][0], tid);

    const int r  = kc >> 2;                   // filter tap 0..8
    const int kw = r % 3;
    const int c0 = (kc & 3) * 32;             // cin sub-block
    const int y  = h + r / 3 - 1;             // always valid in [kcB,kcE)

    // --- B fragments from LDS: lane holds K {8h..8h+7, 16+8h..16+8h+7} of its Cout column
    v16bf bfrag[4];
#pragma unroll
    for (int nf = 0; nf < 4; ++nf) {
      const int o = waveN * 64 + nf * 16 + lan;
      const bf16_t* bp = &Bs[buf][0] + o * BROW + half * 8;
      Q2 q{*reinterpret_cast<const u32x4*>(bp),
           *reinterpret_cast<const u32x4*>(bp + 16)};
      bfrag[nf] = __builtin_bit_cast(v16bf, q);
    }

    // --- A fragments direct from NHWC bf16 (L2-resident); halo cols zeroed
    v16bf afrag[2];
#pragma unroll
    for (int mf = 0; mf < 2; ++mf) {
      const int wr   = waveM * 32 + mf * 16 + lan;      // pixel 0..63
      const int xcol = wr + kw - 1;
      const bool valid = (unsigned)xcol < (unsigned)HW;
      const bf16_t* p = xb +
          ((size_t)(b * HW + y) * HW + (valid ? xcol : 0)) * CIN + c0;
      afrag[mf] = load_a_frag(p, valid, half);
    }

    // --- 8 WMMAs: 16x16x32 bf16 -> f32
#pragma unroll
    for (int mf = 0; mf < 2; ++mf)
#pragma unroll
      for (int nf = 0; nf < 4; ++nf)
        acc[mf][nf] = __builtin_amdgcn_wmma_f32_16x16x32_bf16(
            false, afrag[mf], false, bfrag[nf], (short)0, acc[mf][nf],
            false, false);

    __syncthreads();                          // done reading Bs[buf]
  }

  // --- store: each lane owns 8 consecutive w at fixed Cout; NT (write-once)
#pragma unroll
  for (int mf = 0; mf < 2; ++mf) {
    const int wb = waveM * 32 + mf * 16 + half * 8;     // {0,8,...,56}
    if (wb >= HW) continue;                             // pure pad half
#pragma unroll
    for (int nf = 0; nf < 4; ++nf) {
      const int o = waveN * 64 + nf * 16 + lan;
      float* po = out + ((size_t)(b * COUT + o) * HW + h) * HW + wb;
      __builtin_nontemporal_store(acc[mf][nf], reinterpret_cast<v8f*>(po));
    }
  }
}

// ---------------------------------------------------------------------------
// Host entry
// ---------------------------------------------------------------------------
extern "C" void kernel_launch(void* const* d_in, const int* in_sizes, int n_in,
                              void* d_out, int out_size, void* d_ws,
                              size_t ws_size, hipStream_t stream) {
  const float* x     = (const float*)d_in[0];  // [32,128,56,56]
  const float* dict  = (const float*)d_in[1];  // [256,128,3,3]
  const float* coeff = (const float*)d_in[2];  // [256,8]
  const int*   idx   = (const int*)d_in[3];    // [256,8]
  float* out = (float*)d_out;                  // [32,256,56,56]

  // workspace: Wbf (256*1152*2 = 589,824 B) then xb NHWC bf16 (25,690,112 B)
  bf16_t* wbf = (bf16_t*)d_ws;
  bf16_t* xb  = (bf16_t*)((char*)d_ws + 589824);

  build_weights_kernel<<<COUT, 128, 0, stream>>>(dict, coeff, idx, wbf);
  convert_x_kernel<<<B_N * HW, 128, 0, stream>>>(x, xb);
  lookup_conv_kernel<<<B_N * HW, 256, 0, stream>>>(xb, wbf, out);
}
